// NodeBlock_83631603188045
// MI455X (gfx1250) — matrix operand order repaired and checked by reading, not compiled, not run on hardware
//
#include <hip/hip_runtime.h>
#include <hip/hip_bf16.h>

// ---------------- problem constants (match reference) ----------------
#define N_NODES   50000
#define N_EDGES   1600000
#define EDGE_DIM  64
#define NODE_DIM  128
#define GLOB_DIM  64
#define IN_DIM    256       // EDGE_DIM + NODE_DIM + GLOB_DIM
#define HID_DIM   256
#define OUT_DIM   128
#define N_TILES   (N_NODES / 16)   // 3125, exact
#define WAVES     4                // waves per block in the MLP kernel

// ---------------- vector types (CDNA5 wave32) ----------------
typedef __attribute__((ext_vector_type(16))) __bf16 v16bf;
typedef __attribute__((ext_vector_type(8)))  float  v8f;
typedef __attribute__((ext_vector_type(4)))  float  fv4;   // clang vector: OK for nontemporal builtins

union FragU { uint4 q[2]; v16bf v; };

// float -> bf16, round to nearest even
__device__ __forceinline__ unsigned short f2bf(float f) {
  unsigned int u = __float_as_uint(f);
  unsigned int r = ((u >> 16) & 1u) + 0x7fffu;
  return (unsigned short)((u + r) >> 16);
}

// A-fragment (16x32 bf16, MxK) from an LDS tile stored row-major [16][256].
// ISA layout: lane<16 -> M=lane, elems 0..7 = K kt*32+0..7, elems 8..15 = K kt*32+16..23
//             lane>=16 -> M=lane-16, elems 0..7 = K kt*32+8..15, elems 8..15 = K kt*32+24..31
__device__ __forceinline__ v16bf load_frag_a(const unsigned short* t, int kt, int lane) {
  const unsigned short* p = t + (lane & 15) * 256 + kt * 32 + ((lane >> 4) << 3);
  FragU u;
  u.q[0] = *(const uint4*)(p);        // ds_load_b128
  u.q[1] = *(const uint4*)(p + 16);   // ds_load_b128
  return u.v;
}

// B-fragment (32x16 bf16, KxN) from transposed weights stored [N][K=256] bf16.
// ISA layout: lane L holds column N=L%16; lanes 0-15 K=kt*32+0..7,16..23, lanes 16-31 K=kt*32+8..15,24..31.
__device__ __forceinline__ v16bf load_frag_b(const unsigned short* wt, int nt, int kt, int lane) {
  const unsigned short* p = wt + (nt * 16 + (lane & 15)) * 256 + kt * 32 + ((lane >> 4) << 3);
  FragU u;
  u.q[0] = *(const uint4*)(p);        // global_load_b128 (L2 / WGP$ resident)
  u.q[1] = *(const uint4*)(p + 16);
  return u.v;
}

// ---------------- kernel 1: zero the aggregation buffer ----------------
__global__ void zero_agg_kernel(float* __restrict__ agg) {
  int i = blockIdx.x * 256 + threadIdx.x;
  if (i < N_NODES * EDGE_DIM) agg[i] = 0.0f;
}

// ---------------- kernel 2: transpose + bf16-convert weights ----------------
// w1t: [HID][IN] bf16, w2t: [OUT][HID] bf16
__global__ void prep_weights_kernel(const float* __restrict__ W1, const float* __restrict__ W2,
                                    unsigned short* __restrict__ w1t, unsigned short* __restrict__ w2t) {
  int i = blockIdx.x * 256 + threadIdx.x;
  if (i < HID_DIM * IN_DIM) {
    int h = i / IN_DIM, k = i % IN_DIM;
    w1t[i] = f2bf(W1[k * HID_DIM + h]);
  }
  int j = i - HID_DIM * IN_DIM;
  if (j >= 0 && j < OUT_DIM * HID_DIM) {
    int o = j / HID_DIM, k = j % HID_DIM;
    w2t[j] = f2bf(W2[k * OUT_DIM + o]);
  }
}

// ---------------- kernel 3: edge scatter-add (bandwidth-bound phase) ----------------
// One thread per float4 of an edge. Edge stream is 410 MB read exactly once ->
// non-temporal loads so it does not evict the hot 12.8 MB agg buffer from L2.
__global__ void scatter_add_kernel(const float* __restrict__ edges,
                                   const int* __restrict__ receivers,
                                   float* __restrict__ agg) {
  int t = blockIdx.x * 256 + threadIdx.x;        // quarter-row index: N_EDGES*16 = 25.6M
  if (t >= N_EDGES * (EDGE_DIM / 4)) return;
  int e = t >> 4;                                 // edge id
  int q = (t & 15) << 2;                          // feature offset 0..60 step 4
  fv4 v = __builtin_nontemporal_load((const fv4*)(edges) + t);
  int r = receivers[e];                           // broadcast across 16 consecutive lanes
  float* dst = agg + (long long)r * EDGE_DIM + q;
  atomicAdd(dst + 0, v.x);
  atomicAdd(dst + 1, v.y);
  atomicAdd(dst + 2, v.z);
  atomicAdd(dst + 3, v.w);
}

// ---------------- kernel 4: fused collect + MLP via bf16 WMMA ----------------
__global__ __launch_bounds__(WAVES * 32) void node_mlp_wmma_kernel(
    const float* __restrict__ agg, const float* __restrict__ nodes,
    const float* __restrict__ glob, const float* __restrict__ b1,
    const float* __restrict__ b2,
    const unsigned short* __restrict__ w1t, const unsigned short* __restrict__ w2t,
    float* __restrict__ out) {
  __shared__ unsigned short aAll[WAVES * 16 * IN_DIM];   // 32 KB: staged collected input (bf16)
  __shared__ unsigned short hAll[WAVES * 16 * HID_DIM];  // 32 KB: hidden activations (bf16)

  const int lane = threadIdx.x & 31;
  const int wave = threadIdx.x >> 5;
  const int tile = blockIdx.x * WAVES + wave;
  if (tile >= N_TILES) return;   // wave-uniform exit; no cross-wave LDS sharing

  unsigned short* aT = aAll + wave * 16 * IN_DIM;
  unsigned short* hT = hAll + wave * 16 * HID_DIM;
  const int rowBase = tile * 16;

  // ---- stage collected = [agg | nodes | globals] for 16 rows as bf16 in LDS ----
  // 16 rows x 64 float4-groups; region boundaries (64, 192) are float4-aligned.
  for (int g = lane; g < 16 * 64; g += 32) {
    int r  = g >> 6;
    int c4 = (g & 63) << 2;
    int row = rowBase + r;
    float4 v;
    if (c4 < EDGE_DIM)                 v = *(const float4*)(agg   + row * EDGE_DIM + c4);
    else if (c4 < EDGE_DIM + NODE_DIM) v = *(const float4*)(nodes + row * NODE_DIM + (c4 - EDGE_DIM));
    else                               v = *(const float4*)(glob  + (c4 - EDGE_DIM - NODE_DIM));
    unsigned int lo = (unsigned int)f2bf(v.x) | ((unsigned int)f2bf(v.y) << 16);
    unsigned int hi = (unsigned int)f2bf(v.z) | ((unsigned int)f2bf(v.w) << 16);
    *(uint2*)(aT + r * IN_DIM + c4) = make_uint2(lo, hi);
  }
  // Same-wave LDS ops are in-order; only this wave touches aT/hT.

  const int col = lane & 15;
  const int mhi = (lane >> 4) << 3;   // +8 row offset for lanes 16..31

  // ---- layer 1: h = relu(X @ W1 + b1), X:16x256, W1:256x256 ----
  #pragma unroll
  for (int n1 = 0; n1 < HID_DIM / 16; ++n1) {
    v8f acc = {};
    #pragma unroll
    for (int kt = 0; kt < IN_DIM / 32; ++kt) {
      v16bf a = load_frag_a(aT, kt, lane);
      v16bf b = load_frag_b(w1t, n1, kt, lane);
      acc = __builtin_amdgcn_wmma_f32_16x16x32_bf16(false, a, false, b,
                                                    (short)0, acc, false, false);
    }
    float bias = b1[n1 * 16 + col];
    #pragma unroll
    for (int v = 0; v < 8; ++v) {
      float x = acc[v] + bias;
      x = x > 0.0f ? x : 0.0f;
      hT[(v + mhi) * HID_DIM + n1 * 16 + col] = f2bf(x);
    }
  }

  // ---- layer 2: out = h @ W2 + b2, h:16x256, W2:256x128 ----
  #pragma unroll
  for (int n2 = 0; n2 < OUT_DIM / 16; ++n2) {
    v8f acc = {};
    #pragma unroll
    for (int kt = 0; kt < HID_DIM / 32; ++kt) {
      v16bf a = load_frag_a(hT, kt, lane);
      v16bf b = load_frag_b(w2t, n2, kt, lane);
      acc = __builtin_amdgcn_wmma_f32_16x16x32_bf16(false, a, false, b,
                                                    (short)0, acc, false, false);
    }
    float bias = b2[n2 * 16 + col];
    #pragma unroll
    for (int v = 0; v < 8; ++v) {
      int row = rowBase + v + mhi;
      // result is written once, never re-read on device -> non-temporal store
      __builtin_nontemporal_store(acc[v] + bias,
                                  out + row * OUT_DIM + n2 * 16 + col);
    }
  }
}

// ---------------- host-side launch ----------------
extern "C" void kernel_launch(void* const* d_in, const int* in_sizes, int n_in,
                              void* d_out, int out_size, void* d_ws, size_t ws_size,
                              hipStream_t stream) {
  const float* edges     = (const float*)d_in[0];
  const int*   receivers = (const int*)  d_in[1];
  const float* nodes     = (const float*)d_in[2];
  const float* glob      = (const float*)d_in[3];
  // d_in[4] = num_nodes scalar (compile-time constant here)
  const float* W1 = (const float*)d_in[5];
  const float* b1 = (const float*)d_in[6];
  const float* W2 = (const float*)d_in[7];
  const float* b2 = (const float*)d_in[8];
  float* out = (float*)d_out;

  // workspace layout
  char* ws = (char*)d_ws;
  float* agg = (float*)ws;                                       // 50000*64*4 = 12.8 MB
  size_t aggBytes = (size_t)N_NODES * EDGE_DIM * sizeof(float);  // 32B-aligned
  unsigned short* w1t = (unsigned short*)(ws + aggBytes);        // 256*256 bf16 = 128 KB
  unsigned short* w2t = w1t + (size_t)HID_DIM * IN_DIM;          // 128*256 bf16 = 64 KB

  // 1) zero agg
  {
    int n = N_NODES * EDGE_DIM;
    zero_agg_kernel<<<(n + 255) / 256, 256, 0, stream>>>(agg);
  }
  // 2) transpose + bf16 weights
  {
    int n = HID_DIM * IN_DIM + OUT_DIM * HID_DIM;
    prep_weights_kernel<<<(n + 255) / 256, 256, 0, stream>>>(W1, W2, w1t, w2t);
  }
  // 3) edge scatter-add (float4 per thread, non-temporal edge stream)
  {
    int n = N_EDGES * (EDGE_DIM / 4);          // 25.6M threads
    scatter_add_kernel<<<(n + 255) / 256, 256, 0, stream>>>(edges, receivers, agg);
  }
  // 4) fused collect + MLP (WMMA)
  {
    int blocks = (N_TILES + WAVES - 1) / WAVES;   // 782
    node_mlp_wmma_kernel<<<blocks, WAVES * 32, 0, stream>>>(
        agg, nodes, glob, b1, b2, w1t, w2t, out);
  }
}
